// RDLoss_42520176230472
// MI455X (gfx1250) — matrix operand order repaired and checked by reading, not compiled, not run on hardware
//
#include <hip/hip_runtime.h>
#include <math.h>

#define N 512
#define SIGMA 0.5f
#define EPSL 1e-7f

typedef _Float16 half8 __attribute__((ext_vector_type(8)));
typedef _Float16 v16h  __attribute__((ext_vector_type(16)));
typedef float    v8f   __attribute__((ext_vector_type(8)));

// ---------------------------------------------------------------- init output
__global__ void zero_out_kernel(float* out) {
    if (threadIdx.x == 0 && blockIdx.x == 0) out[0] = 0.0f;
}

// ---------------------------------------- f32 -> f16 copy + exact row norms^2
__global__ void __launch_bounds__(256) cvt_norms_kernel(
    const float* __restrict__ A, _Float16* __restrict__ Ah,
    float* __restrict__ nrm) {
    __shared__ float red[256];
    const int row = blockIdx.x;
    const int tid = threadIdx.x;
    const float* a = A + row * N;
    float s = 0.0f;
    for (int j = tid; j < N; j += 256) {
        float v = a[j];
        Ah[row * N + j] = (_Float16)v;
        s += v * v;
    }
    red[tid] = s;
    __syncthreads();
    for (int off = 128; off > 0; off >>= 1) {
        if (tid < off) red[tid] += red[tid + off];
        __syncthreads();
    }
    if (tid == 0) nrm[row] = red[0];
}

// --------------------------------------------- G = A * A^T via WMMA f16->f32
// One wave32 per 16x16 output tile; K-loop over 512 in steps of 32.
// A-frag (16x32 f16): lane L: row M = L%16; VGPR p in 0..3 holds
//   K = 8*(L/16) + 2p,2p+1 ; VGPR 4..7 holds K = 16 + 8*(L/16) + ...
// B-frag (32x16 f16): lane L: col N = L%16; VGPR p in 0..3 holds
//   K = 8*(L/16) + 2p,2p+1 ; VGPR 4..7 holds K = 16 + 8*(L/16) + ...
__global__ void __launch_bounds__(256) gram_wmma_kernel(
    const _Float16* __restrict__ Ah, float* __restrict__ G) {
    const int lane = threadIdx.x & 31;
    const int wave = threadIdx.x >> 5;
    const int t = blockIdx.x * 8 + wave;     // tile id, 0..1023
    const int tileM = t >> 5;
    const int tileN = t & 31;
    const int mrow  = tileM * 16 + (lane & 15);
    const int ncol  = tileN * 16 + (lane & 15);
    const int abase = (lane >> 4) * 8;       // A-frag K sub-offset
    const int kh    = (lane >> 4) * 8;       // B-frag K sub-offset

    v8f c = {};
    for (int kk = 0; kk < N; kk += 32) {
        half8 a0 = *(const half8*)(Ah + mrow * N + kk + abase);
        half8 a1 = *(const half8*)(Ah + mrow * N + kk + 16 + abase);
        v16h a = __builtin_shufflevector(a0, a1,
                 0,1,2,3,4,5,6,7,8,9,10,11,12,13,14,15);
        half8 b0 = *(const half8*)(Ah + ncol * N + kk + kh);
        half8 b1 = *(const half8*)(Ah + ncol * N + kk + 16 + kh);
        v16h b = __builtin_shufflevector(b0, b1,
                 0,1,2,3,4,5,6,7,8,9,10,11,12,13,14,15);
        // (neg_a, A, neg_b, B, c_mod, C, reuse_a, reuse_b)
        c = __builtin_amdgcn_wmma_f32_16x16x32_f16(
                false, a, false, b, (short)0, c, false, false);
    }
    // C/D layout: VGPR r -> M = r + 8*(lane/16), N = lane%16
    const int gm0 = tileM * 16 + ((lane >> 4) << 3);
    const int gn  = tileN * 16 + (lane & 15);
#pragma unroll
    for (int r = 0; r < 8; ++r) G[(gm0 + r) * N + gn] = c[r];
}

// ------------------ F = -0.5*dist, row softmax; S written in place over G.
// Also stores rowmax and log(sum exp) so F can be recovered: F = logS+lse+mx.
__global__ void __launch_bounds__(256) dist_softmax_kernel(
    const float* __restrict__ nrm, float* __restrict__ G,
    float* __restrict__ mx, float* __restrict__ lse) {
    __shared__ float red[256];
    const int i = blockIdx.x;
    const int tid = threadIdx.x;
    const float ni = nrm[i];

    float f[2];
    const int js[2] = { tid, tid + 256 };
#pragma unroll
    for (int u = 0; u < 2; ++u) {
        const int j = js[u];
        float d2 = ni + nrm[j] - 2.0f * G[i * N + j];
        float dist = (j == i) ? 0.0f : sqrtf(fmaxf(d2, 0.0f));
        f[u] = -SIGMA * dist;
    }
    red[tid] = fmaxf(f[0], f[1]);
    __syncthreads();
    for (int off = 128; off > 0; off >>= 1) {
        if (tid < off) red[tid] = fmaxf(red[tid], red[tid + off]);
        __syncthreads();
    }
    const float rowmax = red[0];
    __syncthreads();

    float e[2];
    float s = 0.0f;
#pragma unroll
    for (int u = 0; u < 2; ++u) { e[u] = expf(f[u] - rowmax); s += e[u]; }
    red[tid] = s;
    __syncthreads();
    for (int off = 128; off > 0; off >>= 1) {
        if (tid < off) red[tid] += red[tid + off];
        __syncthreads();
    }
    const float sum = red[0];
    const float inv = 1.0f / sum;
#pragma unroll
    for (int u = 0; u < 2; ++u) G[i * N + js[u]] = e[u] * inv;
    if (tid == 0) { mx[i] = rowmax; lse[i] = logf(sum); }
}

// ------------------------------ denom mask-sum + loss reduction (per row i).
__global__ void __launch_bounds__(256) denom_loss_kernel(
    const float* __restrict__ S, const float* __restrict__ label,
    const float* __restrict__ mx, const float* __restrict__ lse,
    float* __restrict__ out) {
    __shared__ float Srow[N];
    __shared__ float Rrow[N];
    __shared__ float red[256];
    const int i = blockIdx.x;
    const int tid = threadIdx.x;
    const float li = label[i];
    for (int j = tid; j < N; j += 256) {
        Srow[j] = S[i * N + j];
        Rrow[j] = fabsf(label[j] - li);
    }
    __syncthreads();

    const float base = lse[i] + mx[i];   // F[i,k] = log(S[i,k]) + base
    float partial = 0.0f;
    for (int k = tid; k < N; k += 256) {
        if (k == i) continue;
        const float rk = Rrow[k];
        float den = 0.0f;
#pragma unroll 4
        for (int j = 0; j < N; ++j) {
            if (j != i && Rrow[j] >= rk) den += Srow[j];
        }
        const float Fik = logf(Srow[k]) + base;
        partial += Fik - logf(den + EPSL);
    }
    red[tid] = partial;
    __syncthreads();
    for (int off = 128; off > 0; off >>= 1) {
        if (tid < off) red[tid] += red[tid + off];
        __syncthreads();
    }
    if (tid == 0)
        atomicAdd(out, -red[0] * (1.0f / ((float)N * (float)(N - 1))));
}

// ---------------------------------------------------------------------------
extern "C" void kernel_launch(void* const* d_in, const int* in_sizes, int n_in,
                              void* d_out, int out_size, void* d_ws, size_t ws_size,
                              hipStream_t stream) {
    (void)in_sizes; (void)n_in; (void)out_size; (void)ws_size;
    const float* feature = (const float*)d_in[0];  // [512,512] f32
    const float* label   = (const float*)d_in[1];  // [512,1]  f32
    float* out = (float*)d_out;

    // Workspace layout (~1.51 MB):
    char* ws = (char*)d_ws;
    _Float16* Ah = (_Float16*)ws;                       // 512*512*2 = 512 KB
    float* G     = (float*)(ws + 512 * N * sizeof(_Float16)); // 1 MB (G, then S in place)
    float* nrm   = (float*)((char*)G + (size_t)N * N * sizeof(float));
    float* mx    = nrm + N;
    float* lse   = mx + N;

    zero_out_kernel<<<1, 32, 0, stream>>>(out);
    cvt_norms_kernel<<<N, 256, 0, stream>>>(feature, Ah, nrm);
    gram_wmma_kernel<<<(N / 16) * (N / 16) / 8, 256, 0, stream>>>(Ah, G);
    dist_softmax_kernel<<<N, 256, 0, stream>>>(nrm, G, mx, lse);
    denom_loss_kernel<<<N, 256, 0, stream>>>(G, label, mx, lse, out);
}
